// HookedAttention_32057635897651
// MI455X (gfx1250) — matrix operand order repaired and checked by reading, compiled, not run on hardware
//
#include <hip/hip_runtime.h>
#include <hip/hip_bf16.h>

// ---------------------------------------------------------------------------
// HookedAttention on MI455X (gfx1250): bf16 WMMA everywhere, fp32 accumulate.
// GEMMs use 64x64 register blocking per wave (16 WMMAs per K-step, 4x reuse
// of A/B fragments) to push arithmetic intensity to ~32 FLOP/B from L2.
// ---------------------------------------------------------------------------

typedef __attribute__((ext_vector_type(16))) __bf16 v16bf;
typedef __attribute__((ext_vector_type(8)))  __bf16 v8bf;
typedef __attribute__((ext_vector_type(8)))  float  v8f;

#define N_HEADS 16
#define D_HEAD  64
#define D_MODEL 1024
#define T_SEQ   2048
#define BATCH   2
#define BT      (BATCH * T_SEQ)          // 4096 tokens
#define QKV_DIM (3 * N_HEADS * D_HEAD)   // 3072

static __device__ __forceinline__ v8f wmma_bf16(v16bf a, v16bf b, v8f c) {
    // D = A(16x32 bf16) * B(32x16 bf16) + C(16x16 f32)
    return __builtin_amdgcn_wmma_f32_16x16x32_bf16(
        /*neg_a=*/false, a, /*neg_b=*/false, b,
        /*c_mod=*/(short)0, c, /*reuse_a=*/false, /*reuse_b=*/false);
}

static __device__ __forceinline__ v16bf cat8(v8bf lo, v8bf hi) {
    v16bf r;
#pragma unroll
    for (int i = 0; i < 8; ++i) { r[i] = lo[i]; r[i + 8] = hi[i]; }
    return r;
}

// A-fragment (16x32 bf16, row-major source, row stride `ld` elements):
// lane = row (lane&15); K = (lane>>4)*16 + {0..15}.
static __device__ __forceinline__ v16bf load_afrag(const __bf16* base, int ld,
                                                   int l16, int lh) {
    const __bf16* p = base + (size_t)l16 * ld + lh * 16;
    return cat8(*(const v8bf*)(p), *(const v8bf*)(p + 8));
}

// B-fragment (32x16 bf16) from a K-contiguous (transposed) matrix:
// lane = col (lane&15); K = (lane>>4)*16 + {0..15}.
static __device__ __forceinline__ v16bf load_bfrag(const __bf16* base, int ld,
                                                   int l16, int lh) {
    return *(const v16bf*)(base + (size_t)l16 * ld + lh * 16);
}

// ---------------------------------------------------------------------------
// Stage 0: fp32 -> bf16 conversions (x straight copy; weights transposed so
// WMMA B-fragments become contiguous 32B loads).
// ---------------------------------------------------------------------------
__global__ void ha_cvt_bf16(const float* __restrict__ s, __bf16* __restrict__ d, int n) {
    for (int i = blockIdx.x * blockDim.x + threadIdx.x; i < n; i += gridDim.x * blockDim.x)
        d[i] = (__bf16)s[i];
}

// W[rows][cols] (row-major) -> Wt[cols][rows] bf16
__global__ void ha_cvt_transpose(const float* __restrict__ W, __bf16* __restrict__ Wt,
                                 int rows, int cols) {
    int n = rows * cols;
    for (int i = blockIdx.x * blockDim.x + threadIdx.x; i < n; i += gridDim.x * blockDim.x) {
        int nn = i / rows;      // output row  = original column
        int kk = i % rows;      // output col  = original row (K dim, contiguous)
        Wt[i] = (__bf16)W[kk * cols + nn];
    }
}

// ---------------------------------------------------------------------------
// Stage 1: QKV projection with 64x64 macro-tiles per wave.
// xb[BT,1024] @ WqkvT -> Q [B,H,T,Dh], K [B,H,T,Dh], V transposed [B,H,Dh,T].
// ---------------------------------------------------------------------------
__global__ void __launch_bounds__(128)
ha_qkv_gemm(const __bf16* __restrict__ xb, const __bf16* __restrict__ WqkvT,
            const float* __restrict__ bqkv,
            __bf16* __restrict__ Qb, __bf16* __restrict__ Kb, __bf16* __restrict__ Vt) {
    const int lane = threadIdx.x & 31;
    const int wid  = threadIdx.x >> 5;
    const int tn   = blockIdx.x * 4 + wid;   // 0..47  (64-wide col macro-tile)
    const int tm   = blockIdx.y;             // 0..63  (64-tall row macro-tile)
    const int l16  = lane & 15, lh = lane >> 4;

    const __bf16* abase = xb    + (size_t)(tm * 64) * D_MODEL;
    const __bf16* bbase = WqkvT + (size_t)(tn * 64) * D_MODEL;

    v8f acc[4][4];
#pragma unroll
    for (int i = 0; i < 4; ++i)
#pragma unroll
        for (int j = 0; j < 4; ++j) acc[i][j] = (v8f){};

#pragma unroll 2
    for (int kk = 0; kk < D_MODEL / 32; ++kk) {
        v16bf afr[4], bfr[4];
#pragma unroll
        for (int i = 0; i < 4; ++i)
            afr[i] = load_afrag(abase + (size_t)(i * 16) * D_MODEL + kk * 32,
                                D_MODEL, l16, lh);
#pragma unroll
        for (int j = 0; j < 4; ++j)
            bfr[j] = load_bfrag(bbase + (size_t)(j * 16) * D_MODEL + kk * 32,
                                D_MODEL, l16, lh);
#pragma unroll
        for (int i = 0; i < 4; ++i)
#pragma unroll
            for (int j = 0; j < 4; ++j)
                acc[i][j] = wmma_bf16(afr[i], bfr[j], acc[i][j]);
    }

#pragma unroll
    for (int j = 0; j < 4; ++j) {
        const int c = tn * 64 + j * 16 + l16;
        const float bias = bqkv[c];
        const int which = c >> 10;           // 0=q 1=k 2=v
        const int rem = c & 1023;
        const int h = rem >> 6, d = rem & 63;
#pragma unroll
        for (int i = 0; i < 4; ++i) {
#pragma unroll
            for (int jj = 0; jj < 8; ++jj) {
                int r  = tm * 64 + i * 16 + lh * 8 + jj;   // token row
                int bb = r >> 11, t = r & (T_SEQ - 1);
                __bf16 v = (__bf16)(acc[i][j][jj] + bias);
                if (which == 0)
                    Qb[((size_t)(bb * N_HEADS + h) * T_SEQ + t) * D_HEAD + d] = v;
                else if (which == 1)
                    Kb[((size_t)(bb * N_HEADS + h) * T_SEQ + t) * D_HEAD + d] = v;
                else
                    Vt[((size_t)(bb * N_HEADS + h) * D_HEAD + d) * T_SEQ + t] = v;
            }
        }
    }
}

// ---------------------------------------------------------------------------
// Stage 2: causal flash attention with softmax sink.  One wave per
// (batch, head, 16-query tile).  32-key blocks: 4 WMMAs (scores) +
// LDS online softmax + 4 WMMAs (PV).
// ---------------------------------------------------------------------------
__global__ void __launch_bounds__(32)
ha_attn(const __bf16* __restrict__ Qb, const __bf16* __restrict__ Kb,
        const __bf16* __restrict__ Vt, const float* __restrict__ sink_logit,
        __bf16* __restrict__ z) {
    __shared__ __align__(32) float  Ssc[16][32];
    __shared__ __align__(32) __bf16 Pm[16][32];

    const int qt   = blockIdx.x;          // 0..127
    const int h    = blockIdx.y;          // 0..15
    const int bi   = blockIdx.z;          // 0..1
    const int lane = threadIdx.x;
    const int l16  = lane & 15, lh = lane >> 4;
    const float scale = 0.125f;           // 1/sqrt(64)

    const __bf16* Qh = Qb + (size_t)(bi * N_HEADS + h) * T_SEQ * D_HEAD;
    const __bf16* Kh = Kb + (size_t)(bi * N_HEADS + h) * T_SEQ * D_HEAD;
    const __bf16* Vh = Vt + (size_t)(bi * N_HEADS + h) * D_HEAD * T_SEQ;

    // Q A-fragments for the two 32-wide K-dim chunks (dims 0..31 / 32..63)
    const __bf16* qrow = Qh + (size_t)(qt * 16 + l16) * D_HEAD + lh * 16;
    v16bf qa0 = cat8(*(const v8bf*)(qrow +  0), *(const v8bf*)(qrow +  8));
    v16bf qa1 = cat8(*(const v8bf*)(qrow + 32), *(const v8bf*)(qrow + 40));

    v8f o0 = {}, o1 = {}, o2 = {}, o3 = {};
    float m = -1e30f, l = 0.f, alpha = 0.f;
    const int qg  = qt * 16 + lane;               // query row owned by lanes 0..15
    const int nkb = (qt * 16 + 15) / 32 + 1;      // key blocks to visit (causal)

    for (int kb = 0; kb < nkb; ++kb) {
        const int key0 = kb * 32;

        // ---- scores: S[16 x 32] = Q (16x64) x K^T block ----
        v8f s0 = {}, s1 = {};
        {
            const __bf16* kr0 = Kh + (size_t)(key0 + l16) * D_HEAD + lh * 16;
            const __bf16* kr1 = Kh + (size_t)(key0 + 16 + l16) * D_HEAD + lh * 16;
            s0 = wmma_bf16(qa0, *(const v16bf*)(kr0),      s0);
            s0 = wmma_bf16(qa1, *(const v16bf*)(kr0 + 32), s0);
            s1 = wmma_bf16(qa0, *(const v16bf*)(kr1),      s1);
            s1 = wmma_bf16(qa1, *(const v16bf*)(kr1 + 32), s1);
        }
        if (kb + 1 < nkb) {  // prefetch next K/V blocks (-> global_prefetch_b8)
            __builtin_prefetch((const void*)(Kh + (size_t)(key0 + 32 + l16) * D_HEAD), 0, 1);
            __builtin_prefetch((const void*)(Vh + (size_t)l16 * T_SEQ + key0 + 32), 0, 1);
        }

        // stash C-fragment scores to LDS: row = lh*8+j, col = l16 (+16 for s1)
#pragma unroll
        for (int j = 0; j < 8; ++j) {
            Ssc[lh * 8 + j][l16]      = s0[j];
            Ssc[lh * 8 + j][16 + l16] = s1[j];
        }
        __syncthreads();

        // ---- online softmax: lanes 0..15 each own one query row ----
        if (lane < 16) {
            float mloc = -1e30f;
#pragma unroll
            for (int c2 = 0; c2 < 32; ++c2) {
                float v = (key0 + c2 <= qg) ? Ssc[lane][c2] * scale : -1e30f;
                mloc = fmaxf(mloc, v);
            }
            float mnew = fmaxf(m, mloc);
            alpha = __expf(m - mnew);
            float ls = 0.f;
#pragma unroll
            for (int c2 = 0; c2 < 32; ++c2) {
                float v = (key0 + c2 <= qg) ? Ssc[lane][c2] * scale : -1e30f;
                float p = __expf(v - mnew);
                ls += p;
                Pm[lane][c2] = (__bf16)p;
            }
            l = l * alpha + ls;
            m = mnew;
        }
        __syncthreads();

        // rescale running output accumulators by per-row alpha
#pragma unroll
        for (int j = 0; j < 8; ++j) {
            float af = __shfl(alpha, lh * 8 + j, 32);
            o0[j] *= af; o1[j] *= af; o2[j] *= af; o3[j] *= af;
        }

        // ---- PV: O[16x64] += P(16x32 bf16) x V(32x64) ----
        v16bf pa = cat8(*(const v8bf*)(&Pm[l16][lh * 16]),
                        *(const v8bf*)(&Pm[l16][lh * 16 + 8]));
        const __bf16* vb = Vh + (size_t)l16 * T_SEQ + key0 + lh * 16;
        o0 = wmma_bf16(pa, *(const v16bf*)(vb + (size_t)(0 * 16) * T_SEQ), o0);
        o1 = wmma_bf16(pa, *(const v16bf*)(vb + (size_t)(1 * 16) * T_SEQ), o1);
        o2 = wmma_bf16(pa, *(const v16bf*)(vb + (size_t)(2 * 16) * T_SEQ), o2);
        o3 = wmma_bf16(pa, *(const v16bf*)(vb + (size_t)(3 * 16) * T_SEQ), o3);
        __syncthreads();
    }

    // finalize: denominator includes the sink column exp(sink - m); v_sink = 0
    float rinv = 0.f;
    if (lane < 16) rinv = 1.0f / (l + __expf(sink_logit[h] - m));

#pragma unroll
    for (int j = 0; j < 8; ++j) {
        float rf = __shfl(rinv, lh * 8 + j, 32);
        int row = bi * T_SEQ + qt * 16 + lh * 8 + j;
        __bf16* zp = z + (size_t)row * D_MODEL + h * D_HEAD + l16;
        zp[0]  = (__bf16)(o0[j] * rf);
        zp[16] = (__bf16)(o1[j] * rf);
        zp[32] = (__bf16)(o2[j] * rf);
        zp[48] = (__bf16)(o3[j] * rf);
    }
}

// ---------------------------------------------------------------------------
// Stage 3: output projection z[BT,1024] @ W_O + b_O -> fp32 out.
// 64x64 macro-tiles per wave, same blocking as stage 1.
// ---------------------------------------------------------------------------
__global__ void __launch_bounds__(128)
ha_out_gemm(const __bf16* __restrict__ zb, const __bf16* __restrict__ WoT,
            const float* __restrict__ bo, float* __restrict__ out) {
    const int lane = threadIdx.x & 31;
    const int wid  = threadIdx.x >> 5;
    const int tn   = blockIdx.x * 4 + wid;   // 0..15
    const int tm   = blockIdx.y;             // 0..63
    const int l16  = lane & 15, lh = lane >> 4;

    const __bf16* abase = zb  + (size_t)(tm * 64) * D_MODEL;
    const __bf16* bbase = WoT + (size_t)(tn * 64) * D_MODEL;

    v8f acc[4][4];
#pragma unroll
    for (int i = 0; i < 4; ++i)
#pragma unroll
        for (int j = 0; j < 4; ++j) acc[i][j] = (v8f){};

#pragma unroll 2
    for (int kk = 0; kk < D_MODEL / 32; ++kk) {
        v16bf afr[4], bfr[4];
#pragma unroll
        for (int i = 0; i < 4; ++i)
            afr[i] = load_afrag(abase + (size_t)(i * 16) * D_MODEL + kk * 32,
                                D_MODEL, l16, lh);
#pragma unroll
        for (int j = 0; j < 4; ++j)
            bfr[j] = load_bfrag(bbase + (size_t)(j * 16) * D_MODEL + kk * 32,
                                D_MODEL, l16, lh);
#pragma unroll
        for (int i = 0; i < 4; ++i)
#pragma unroll
            for (int j = 0; j < 4; ++j)
                acc[i][j] = wmma_bf16(afr[i], bfr[j], acc[i][j]);
    }

#pragma unroll
    for (int j = 0; j < 4; ++j) {
        const int c = tn * 64 + j * 16 + l16;
        const float bias = bo[c];
#pragma unroll
        for (int i = 0; i < 4; ++i)
#pragma unroll
            for (int jj = 0; jj < 8; ++jj)
                out[(size_t)(tm * 64 + i * 16 + lh * 8 + jj) * D_MODEL + c] =
                    acc[i][j][jj] + bias;
    }
}

// ---------------------------------------------------------------------------
extern "C" void kernel_launch(void* const* d_in, const int* in_sizes, int n_in,
                              void* d_out, int out_size, void* d_ws, size_t ws_size,
                              hipStream_t stream) {
    const float* x    = (const float*)d_in[0];
    const float* Wqkv = (const float*)d_in[1];
    const float* bqkv = (const float*)d_in[2];
    const float* Wo   = (const float*)d_in[3];
    const float* bo   = (const float*)d_in[4];
    const float* sink = (const float*)d_in[5];
    float* out = (float*)d_out;

    // workspace carve-out (256B-aligned)
    char* ws = (char*)d_ws;
    size_t off = 0;
    auto carve = [&](size_t bytes) -> void* {
        void* p = ws + off;
        off += (bytes + 255) & ~(size_t)255;
        return p;
    };
    __bf16* xb    = (__bf16*)carve((size_t)BT * D_MODEL * 2);
    __bf16* WqkvT = (__bf16*)carve((size_t)QKV_DIM * D_MODEL * 2);
    __bf16* WoT   = (__bf16*)carve((size_t)D_MODEL * D_MODEL * 2);
    __bf16* Qb    = (__bf16*)carve((size_t)BATCH * N_HEADS * T_SEQ * D_HEAD * 2);
    __bf16* Kb    = (__bf16*)carve((size_t)BATCH * N_HEADS * T_SEQ * D_HEAD * 2);
    __bf16* Vt    = (__bf16*)carve((size_t)BATCH * N_HEADS * D_HEAD * T_SEQ * 2);
    __bf16* zb    = (__bf16*)carve((size_t)BT * D_MODEL * 2);
    (void)ws_size; (void)in_sizes; (void)n_in; (void)out_size;

    // Stage 0: precision conversion / weight transposes
    ha_cvt_bf16<<<2048, 256, 0, stream>>>(x, xb, BT * D_MODEL);
    ha_cvt_transpose<<<2048, 256, 0, stream>>>(Wqkv, WqkvT, D_MODEL, QKV_DIM);
    ha_cvt_transpose<<<1024, 256, 0, stream>>>(Wo, WoT, D_MODEL, D_MODEL);

    // Stage 1: QKV projection (64 x 48 macro-tiles of 64x64, 4 waves/block)
    ha_qkv_gemm<<<dim3(QKV_DIM / 256, BT / 64), 128, 0, stream>>>(
        xb, WqkvT, bqkv, Qb, Kb, Vt);

    // Stage 2: causal attention + sink (one wave per (b,h,q-tile))
    ha_attn<<<dim3(T_SEQ / 16, N_HEADS, BATCH), 32, 0, stream>>>(
        Qb, Kb, Vt, sink, zb);

    // Stage 3: output projection (64 x 16 macro-tiles)
    ha_out_gemm<<<dim3(D_MODEL / 256, BT / 64), 128, 0, stream>>>(zb, WoT, bo, out);
}
